// GaussianRasterizer_74809740362338
// MI455X (gfx1250) — matrix operand order repaired and checked by reading, not compiled, not run on hardware
//
#include <hip/hip_runtime.h>
#include <hip/hip_bf16.h>
#include <math.h>

typedef __attribute__((ext_vector_type(16))) _Float16 v16h;
typedef __attribute__((ext_vector_type(8)))  float    v8f;

#define N_G   2048
#define IMG_H 128
#define IMG_W 128
#define FXc   100.0f
#define FYc   100.0f
#define CXc   (IMG_W / 2.0f)
#define CYc   (IMG_H / 2.0f)
#define NEARc 0.01f
#define FARc  100.0f
#define REC   12   // floats per record: mx,my,ia,ib | id,op,r,g | b,0,0,0 (3x float4)
#define CHUNK 32
#define CHUNK_F (CHUNK * REC)          // 384 floats = 1536 B per chunk

// ---------------------------------------------------------------------------
// CDNA5 async global->LDS copy (ASYNCcnt-tracked DMA), 16B per enabled lane.
// lds_off = wave-relative LDS byte address (low 32 bits of generic pointer,
// per ISA aperture rules). gaddr = 64-bit global address.
// ---------------------------------------------------------------------------
__device__ __forceinline__ void async_to_lds_b128(unsigned int lds_off,
                                                  unsigned long long gaddr) {
    asm volatile("global_load_async_to_lds_b128 %0, %1, off"
                 :
                 : "v"(lds_off), "v"(gaddr)
                 : "memory");
}
__device__ __forceinline__ void wait_asynccnt0() {
    asm volatile("s_wait_asynccnt 0x0" ::: "memory");
}

// ---------------------------------------------------------------------------
// Kernel 1: project all gaussians, bitonic-sort by depth in LDS, emit sorted
// compact records to workspace. One workgroup of 1024 threads (32 waves).
// ---------------------------------------------------------------------------
__global__ __launch_bounds__(1024)
void k_project_sort(const float* __restrict__ pos,
                    const float* __restrict__ covs,
                    const float* __restrict__ ops,
                    const float* __restrict__ cols,
                    const float* __restrict__ camp,
                    const float* __restrict__ camq,
                    float* __restrict__ tmp,   // unsorted records [N_G*REC]
                    float* __restrict__ rec)   // sorted records   [N_G*REC]
{
    __shared__ float s_key[N_G];
    __shared__ int   s_idx[N_G];

    float qw = camq[0], qx = camq[1], qy = camq[2], qz = camq[3];
    float qn = rsqrtf(qw*qw + qx*qx + qy*qy + qz*qz);
    qw *= qn; qx *= qn; qy *= qn; qz *= qn;
    float R00 = 1.f - 2.f*(qy*qy + qz*qz), R01 = 2.f*(qx*qy - qz*qw), R02 = 2.f*(qx*qz + qy*qw);
    float R10 = 2.f*(qx*qy + qz*qw), R11 = 1.f - 2.f*(qx*qx + qz*qz), R12 = 2.f*(qy*qz - qx*qw);
    float R20 = 2.f*(qx*qz - qy*qw), R21 = 2.f*(qy*qz + qx*qw), R22 = 1.f - 2.f*(qx*qx + qy*qy);
    float ccx = camp[0], ccy = camp[1], ccz = camp[2];
    float tx = -(R00*ccx + R01*ccy + R02*ccz);
    float ty = -(R10*ccx + R11*ccy + R12*ccz);
    float tz = -(R20*ccx + R21*ccy + R22*ccz);

    for (int g = threadIdx.x; g < N_G; g += blockDim.x) {
        float px = pos[g*3 + 0], py = pos[g*3 + 1], pz = pos[g*3 + 2];
        float x = R00*px + R01*py + R02*pz + tx;
        float y = R10*px + R11*py + R12*pz + ty;
        float z = R20*px + R21*py + R22*pz + tz;
        float depth = -z;
        bool valid = (depth > NEARc) & (depth < FARc) &
                     (fabsf(x) < depth * (IMG_W / (2.f*FXc))) &
                     (fabsf(y) < depth * (IMG_H / (2.f*FYc)));
        float iz = 1.f / z;
        float mx = FXc * x * iz + CXc;
        float my = FYc * y * iz + CYc;
        float j00 = FXc * iz,  j02 = -FXc * x * iz * iz;
        float j11 = FYc * iz,  j12 = -FYc * y * iz * iz;
        const float* S = covs + g*9;
        float m00 = j00*S[0] + j02*S[6], m01 = j00*S[1] + j02*S[7], m02 = j00*S[2] + j02*S[8];
        float m11 = j11*S[4] + j12*S[7], m12 = j11*S[5] + j12*S[8];
        float a = m00*j00 + m02*j02 + 1e-6f;
        float b = m01*j11 + m02*j12;
        float d = m11*j11 + m12*j12 + 1e-6f;
        float idet = 1.f / (a*d - b*b);
        float ia = d*idet, ib = -b*idet, idd = a*idet;
        float o = ops[g];
        if (!valid) { mx = 0.f; my = 0.f; ia = 0.f; ib = 0.f; idd = 0.f; o = 0.f; }
        float4* T4 = (float4*)(tmp + g*REC);
        T4[0] = make_float4(mx, my, ia, ib);
        T4[1] = make_float4(idd, o, cols[g*3 + 0], cols[g*3 + 1]);
        T4[2] = make_float4(cols[g*3 + 2], 0.f, 0.f, 0.f);
        s_key[g] = valid ? depth : INFINITY;
        s_idx[g] = g;
    }
    __syncthreads();

    // bitonic sort ascending on (key,idx): N/2 = 1024 compare-exchanges/pass
    const int t = threadIdx.x;
    for (int k = 2; k <= N_G; k <<= 1) {
        for (int j = k >> 1; j > 0; j >>= 1) {
            int i   = ((t & ~(j - 1)) << 1) | (t & (j - 1));
            int ixj = i | j;
            bool up = ((i & k) == 0);
            float ki = s_key[i], kj = s_key[ixj];
            if ((ki > kj) == up) {
                s_key[i] = kj; s_key[ixj] = ki;
                int tmpi = s_idx[i]; s_idx[i] = s_idx[ixj]; s_idx[ixj] = tmpi;
            }
            __syncthreads();
        }
    }

    // gather records into sorted order (3x float4 per record)
    for (int s = threadIdx.x; s < N_G; s += blockDim.x) {
        int src = s_idx[s];
        const float4* s4 = (const float4*)(tmp + src*REC);
        float4*       d4 = (float4*)(rec + s*REC);
        d4[0] = s4[0]; d4[1] = s4[1]; d4[2] = s4[2];
    }
}

// ---------------------------------------------------------------------------
// Kernel 2: rasterize. 256 threads = 8 waves, each wave owns 16 pixels.
// Double-buffered async global->LDS staging of 32-splat chunks; per chunk:
// alphas + split-prefix transmittance (one shfl_xor(16) pair), weighted color
// accumulation via v_wmma_f32_16x16x32_f16.
// ---------------------------------------------------------------------------
__global__ __launch_bounds__(256)
void k_raster(const float* __restrict__ rec, float* __restrict__ out)
{
    __shared__ float sg[2][CHUNK_F];

    const int tid  = threadIdx.x;
    const int lane = tid & 31;
    const int wave = tid >> 5;
    const int h    = lane >> 4;   // wave half
    const int n    = lane & 15;   // matrix column (pixel row M during compute)
    const int ch   = (n < 3) ? n : 3;   // color channel; field 6+3 is the 0 pad
    const int pixBase = blockIdx.x * 128 + wave * 16;
    const int pix  = pixBase + n;
    const float fpx = (float)(pix & (IMG_W - 1));
    const float fpy = (float)(pix >> 7);

    // prologue: async-stage chunk 0 into buffer 0 (96 lanes x 16B)
    if (tid < CHUNK_F / 4) {
        async_to_lds_b128((unsigned int)(uintptr_t)&sg[0][tid * 4],
                          (unsigned long long)(uintptr_t)(rec + tid * 4));
    }
    wait_asynccnt0();
    __syncthreads();

    float T  = 1.0f;
    v8f  acc = {};

    for (int c = 0; c < N_G / CHUNK; ++c) {
        const int cur = c & 1;
        // prefetch next chunk into the other buffer while we compute
        if (c + 1 < N_G / CHUNK && tid < CHUNK_F / 4) {
            async_to_lds_b128((unsigned int)(uintptr_t)&sg[cur ^ 1][tid * 4],
                              (unsigned long long)(uintptr_t)(rec + (c + 1) * CHUNK_F + tid * 4));
        }
        const float* S = sg[cur];

        // B matrix (colors): lane column n, K = 8h + j (j<8), 16 + 8h + (j-8) (j>=8)  (branchless via pad)
        v16h bmat;
        #pragma unroll
        for (int j = 0; j < 16; ++j) {
            int k = (j < 8) ? (8 * h + j) : (16 + 8 * h + (j - 8));
            bmat[j] = (_Float16)S[k * REC + 6 + ch];
        }

        float w[16];
        float tA = 1.f, tB = 1.f;
        // stage A: K = 8h + j  (A-matrix per-lane elements 0..7)
        #pragma unroll
        for (int j = 0; j < 8; ++j) {
            int k = 8 * h + j;
            const float4 g0 = *(const float4*)(S + k * REC);      // mx,my,ia,ib
            const float2 g1 = *(const float2*)(S + k * REC + 4);  // id,op
            float dx = fpx - g0.x, dy = fpy - g0.y;
            float maha = g0.z*dx*dx + 2.f*g0.w*dx*dy + g1.x*dy*dy;
            float al = fminf(g1.y * __expf(-0.5f * maha), 0.99f);
            w[j] = al * tA;
            tA *= (1.f - al);
        }
        // stage B: K = 16 + 8h + j  (elements 8..15)
        #pragma unroll
        for (int j = 0; j < 8; ++j) {
            int k = 16 + 8 * h + j;
            const float4 g0 = *(const float4*)(S + k * REC);
            const float2 g1 = *(const float2*)(S + k * REC + 4);
            float dx = fpx - g0.x, dy = fpy - g0.y;
            float maha = g0.z*dx*dx + 2.f*g0.w*dx*dy + g1.x*dy*dy;
            float al = fminf(g1.y * __expf(-0.5f * maha), 0.99f);
            w[8 + j] = al * tB;
            tB *= (1.f - al);
        }
        // exchange stage totals between wave halves (same pixel, partner K set)
        float tA_p = __shfl_xor(tA, 16, 32);
        float tB_p = __shfl_xor(tB, 16, 32);
        float fA = T * (h ? tA_p : 1.f);
        float fB = T * tA * tA_p * (h ? tB_p : 1.f);

        v16h amat;
        #pragma unroll
        for (int j = 0; j < 8; ++j) {
            amat[j]     = (_Float16)(w[j]     * fA);
            amat[8 + j] = (_Float16)(w[8 + j] * fB);
        }

        acc = __builtin_amdgcn_wmma_f32_16x16x32_f16(
                  false, amat, false, bmat, (short)0, acc, false, false);

        T = T * tA * tA_p * tB * tB_p;

        wait_asynccnt0();   // next chunk's DMA has landed
        __syncthreads();
    }

    // D layout: VGPR r -> row M = r + 8h (pixel), lane column n -> channel
    if (n < 3) {
        #pragma unroll
        for (int r = 0; r < 8; ++r) {
            int p = pixBase + r + 8 * h;
            out[p * 3 + n] = acc[r];
        }
    }
}

extern "C" void kernel_launch(void* const* d_in, const int* in_sizes, int n_in,
                              void* d_out, int out_size, void* d_ws, size_t ws_size,
                              hipStream_t stream) {
    (void)in_sizes; (void)n_in; (void)out_size; (void)ws_size;
    const float* pos  = (const float*)d_in[0];
    const float* covs = (const float*)d_in[1];
    const float* ops  = (const float*)d_in[2];
    const float* cols = (const float*)d_in[3];
    const float* camp = (const float*)d_in[4];
    const float* camq = (const float*)d_in[5];
    float* rec = (float*)d_ws;                    // sorted records
    float* tmp = rec + (size_t)N_G * REC;         // unsorted scratch

    k_project_sort<<<1, 1024, 0, stream>>>(pos, covs, ops, cols, camp, camq, tmp, rec);
    k_raster<<<(IMG_H * IMG_W) / 128, 256, 0, stream>>>(rec, (float*)d_out);
}
